// SelfAttention_54992761258372
// MI455X (gfx1250) — matrix operand order repaired and checked by reading, not compiled, not run on hardware
//
#include <hip/hip_runtime.h>
#include <hip/hip_bf16.h>

#define DMODEL 1024
#define SEQ    2048
#define NBATCH 2
#define NHEAD  16
#define HDIM   64
#define MROWS  (NBATCH * SEQ)   // 4096

typedef __attribute__((ext_vector_type(16))) __bf16 v16bf;
typedef __attribute__((ext_vector_type(8)))  float  v8f;
typedef unsigned int u32x4 __attribute__((ext_vector_type(4)));
typedef int          i32x8 __attribute__((ext_vector_type(8)));
typedef int          i32x4 __attribute__((ext_vector_type(4)));

union Frag {
  v16bf v;
  uint4 u[2];
};

__device__ __forceinline__ unsigned short f2bf(float f) {
  union { float f; unsigned int u; } x;
  x.f = f;
  unsigned int r = x.u + 0x7FFFu + ((x.u >> 16) & 1u);
  return (unsigned short)(r >> 16);
}

__device__ __forceinline__ v8f wmma_bf16(v16bf a, v16bf b, v8f c) {
  return __builtin_amdgcn_wmma_f32_16x16x32_bf16(false, a, false, b,
                                                 (short)0, c, false, false);
}

// A-matrix fragment (16x32 bf16, MxK): lane holds row (lane&15);
// elements e<8 -> K = 8*(lane>=16)+e, e>=8 -> K = 8*(lane>=16)+8+e.
__device__ __forceinline__ v16bf load_a(const unsigned short* __restrict__ p,
                                        int ld, int row0, int k0, int lane) {
  const unsigned short* q =
      p + (size_t)(row0 + (lane & 15)) * ld + k0 + ((lane >> 4) << 3);
  Frag f;
  f.u[0] = *(const uint4*)(q);
  f.u[1] = *(const uint4*)(q + 16);
  return f.v;
}

// B-matrix fragment (32x16 bf16, KxN): lane holds column (lane&15);
// elements e<8 -> K = 8*(lane>=16)+e, e>=8 -> K = 8*(lane>=16)+8+e. `p` holds B
// column-major: row n of p is column n of B.
__device__ __forceinline__ v16bf load_b(const unsigned short* __restrict__ p,
                                        int ld, int col0, int k0, int lane) {
  const unsigned short* q =
      p + (size_t)(col0 + (lane & 15)) * ld + k0 + ((lane >> 4) << 3);
  Frag f;
  f.u[0] = *(const uint4*)(q);
  f.u[1] = *(const uint4*)(q + 16);
  return f.v;
}

// LDS variants with padded row stride of 72 elements (32+4 dwords).
#define LDSLD 72
__device__ __forceinline__ v16bf lds_a(const unsigned short* s, int row0,
                                       int k0, int lane) {
  const unsigned short* q =
      s + (row0 + (lane & 15)) * LDSLD + k0 + ((lane >> 4) << 3);
  Frag f;
  f.u[0] = *(const uint4*)(q);
  f.u[1] = *(const uint4*)(q + 16);
  return f.v;
}
__device__ __forceinline__ v16bf lds_b(const unsigned short* s, int row0,
                                       int k0, int lane) {
  const unsigned short* q =
      s + (row0 + (lane & 15)) * LDSLD + k0 + ((lane >> 4) << 3);
  Frag f;
  f.u[0] = *(const uint4*)(q);
  f.u[1] = *(const uint4*)(q + 16);
  return f.v;
}

// ---------------- Tensor Data Mover: 2D tile -> LDS ----------------
// D# per CDNA5 ISA 8.3/8.4. Row padding: every 32 DWORDs (64 bf16 = one tile
// row) insert 4 DWORDs -> LDS row stride 72 elements, bank-conflict free.
// clang-23 toolchain builtin: (u32x4, i32x8, i32x4, i32x4, i32x8, i32 cpol).
__device__ __forceinline__ void tdm_load_2d(unsigned lds_byte_off,
                                            const unsigned short* gp,
                                            unsigned tensor_d0,
                                            unsigned tensor_d1,
                                            unsigned long long stride0,
                                            unsigned tile_d0,
                                            unsigned tile_d1) {
  unsigned long long ga = (unsigned long long)(size_t)gp;
  u32x4 g0;
  g0[0] = 1u;                                     // count=1, user descriptor
  g0[1] = lds_byte_off;                           // lds_addr
  g0[2] = (unsigned)(ga & 0xFFFFFFFFu);           // global_addr[31:0]
  g0[3] = (unsigned)((ga >> 32) & 0x1FFFFFFu)     // global_addr[56:32]
          | (2u << 30);                           // type=2 (image)
  i32x8 g1;
  g1[0] = (int)((1u << 16)      // data_size = 2 bytes
                | (1u << 20)    // pad_enable
                | (4u << 22)    // pad_interval: 32 DWORDs
                | (3u << 25));  // pad_amount:   4 DWORDs
  g1[1] = (int)((tensor_d0 & 0xFFFFu) << 16);
  g1[2] = (int)(((tensor_d0 >> 16) & 0xFFFFu) | ((tensor_d1 & 0xFFFFu) << 16));
  g1[3] = (int)(((tensor_d1 >> 16) & 0xFFFFu) | ((tile_d0 & 0xFFFFu) << 16));
  g1[4] = (int)(tile_d1 & 0xFFFFu);               // tile_dim1 (tile_dim2=0)
  g1[5] = (int)(unsigned)(stride0 & 0xFFFFFFFFu); // tensor_dim0_stride lo
  g1[6] = (int)(unsigned)((stride0 >> 32) & 0xFFFFu);
  g1[7] = 0;
  i32x4 z4 = {0, 0, 0, 0};
  i32x8 z8 = {0, 0, 0, 0, 0, 0, 0, 0};
  __builtin_amdgcn_tensor_load_to_lds(g0, g1, z4, z4, z8, 0);
}

// ---------------- prep kernels ----------------

__global__ void wtrans_kernel(const float* __restrict__ w,
                              unsigned short* __restrict__ wt) {
  __shared__ float tile[32][33];
  int n = blockIdx.x * 32 + threadIdx.x;
  int k0 = blockIdx.y * 32;
  for (int i = threadIdx.y; i < 32; i += 8)
    tile[i][threadIdx.x] = w[(size_t)(k0 + i) * DMODEL + n];
  __syncthreads();
  int k = k0 + threadIdx.x;
  int n0 = blockIdx.x * 32;
  for (int i = threadIdx.y; i < 32; i += 8)
    wt[(size_t)(n0 + i) * DMODEL + k] = f2bf(tile[threadIdx.x][i]);
}

__global__ void cvt_bf16_kernel(const float* __restrict__ x,
                                unsigned short* __restrict__ xb, int n) {
  for (int i = blockIdx.x * blockDim.x + threadIdx.x; i < n;
       i += gridDim.x * blockDim.x)
    xb[i] = f2bf(x[i]);
}

// ---------------- TDM-staged GEMM ----------------
// C[m][n] = sum_k A[m][k] * B[k][n] + bias[n]; A,B^T bf16 row-major, K=N=1024.
// Block: 8 waves = 4(M) x 2(N) wave-tiles of 16x64 -> block tile 64x128.
// K staged in chunks of 64 through double-buffered LDS via TDM, wave 0 driving.
// All 10 fragments of a stage are loaded before the 8-WMMA burst so the matrix
// pipe sees back-to-back WMMAs behind a single DS wait.
// mode 0: store bf16 [B,H,T,Hd]; mode 1: store bf16 [B,H,Hd,T];
// mode 2: store f32 [M,D] + bias + residual.
__global__ void __launch_bounds__(256)
gemm_tdm_kernel(const unsigned short* __restrict__ amat,
                const unsigned short* __restrict__ bmat,
                const float* __restrict__ bias,
                const float* __restrict__ xres,
                unsigned short* __restrict__ out_us,
                float* __restrict__ out_f, int mode) {
  __shared__ __align__(16) unsigned short aslab[2][64 * LDSLD];
  __shared__ __align__(16) unsigned short bslab[2][128 * LDSLD];
  int lane = threadIdx.x & 31;
  int wave = threadIdx.x >> 5;
  int blk_m = blockIdx.x * 64;
  int blk_n = blockIdx.y * 128;
  int m0r = (wave & 3) * 16;
  int n0r = (wave >> 2) * 64;
  v8f acc[4] = {};

  if (wave == 0) {  // prologue: stage 0 (k = 0)
    tdm_load_2d((unsigned)(size_t)&aslab[0][0], amat + (size_t)blk_m * DMODEL,
                DMODEL, MROWS, DMODEL, 64, 64);
    tdm_load_2d((unsigned)(size_t)&bslab[0][0], bmat + (size_t)blk_n * DMODEL,
                DMODEL, DMODEL, DMODEL, 64, 128);
  }
#pragma unroll 1
  for (int s = 0; s < DMODEL / 64; ++s) {
    int cur = s & 1;
    if (wave == 0) {
      if (s + 1 < DMODEL / 64) {
        int k = (s + 1) * 64;
        tdm_load_2d((unsigned)(size_t)&aslab[cur ^ 1][0],
                    amat + (size_t)blk_m * DMODEL + k, DMODEL, MROWS, DMODEL,
                    64, 64);
        tdm_load_2d((unsigned)(size_t)&bslab[cur ^ 1][0],
                    bmat + (size_t)blk_n * DMODEL + k, DMODEL, DMODEL, DMODEL,
                    64, 128);
        __builtin_amdgcn_s_wait_tensorcnt(2);  // stage s complete (in-order)
      } else {
        __builtin_amdgcn_s_wait_tensorcnt(0);
      }
    }
    __syncthreads();  // publish stage s to all waves
    const unsigned short* as = &aslab[cur][0];
    const unsigned short* bs = &bslab[cur][0];
    // hoist every fragment of this stage, then burst the WMMAs
    v16bf a0 = lds_a(as, m0r, 0, lane);
    v16bf a1 = lds_a(as, m0r, 32, lane);
    v16bf b00 = lds_b(bs, n0r + 0,  0,  lane);
    v16bf b01 = lds_b(bs, n0r + 16, 0,  lane);
    v16bf b02 = lds_b(bs, n0r + 32, 0,  lane);
    v16bf b03 = lds_b(bs, n0r + 48, 0,  lane);
    v16bf b10 = lds_b(bs, n0r + 0,  32, lane);
    v16bf b11 = lds_b(bs, n0r + 16, 32, lane);
    v16bf b12 = lds_b(bs, n0r + 32, 32, lane);
    v16bf b13 = lds_b(bs, n0r + 48, 32, lane);
    acc[0] = wmma_bf16(a0, b00, acc[0]);
    acc[1] = wmma_bf16(a0, b01, acc[1]);
    acc[2] = wmma_bf16(a0, b02, acc[2]);
    acc[3] = wmma_bf16(a0, b03, acc[3]);
    acc[0] = wmma_bf16(a1, b10, acc[0]);
    acc[1] = wmma_bf16(a1, b11, acc[1]);
    acc[2] = wmma_bf16(a1, b12, acc[2]);
    acc[3] = wmma_bf16(a1, b13, acc[3]);
    __syncthreads();  // all waves done with stage s before it is overwritten
  }

  int rowadd = (lane >> 4) << 3;
  int col = lane & 15;
#pragma unroll
  for (int c = 0; c < 4; ++c) {
    int n = blk_n + n0r + c * 16 + col;
    float bv = bias[n];
    int h = n >> 6, hd = n & (HDIM - 1);
#pragma unroll
    for (int j = 0; j < 8; ++j) {
      int m = blk_m + m0r + rowadd + j;
      float val = acc[c][j] + bv;
      if (mode == 2) {
        size_t idx = (size_t)m * DMODEL + n;
        out_f[idx] = val + xres[idx];
      } else {
        int b = m >> 11, t = m & (SEQ - 1);
        size_t idx =
            (mode == 0)
                ? ((size_t)((b * NHEAD + h) * SEQ + t) * HDIM + hd)
                : ((size_t)((b * NHEAD + h) * HDIM + hd) * SEQ + t);
        out_us[idx] = f2bf(val);
      }
    }
  }
}

// ---------------- flash attention ----------------
// One wave per 16-row Q tile; 32 keys per step, online softmax, K-fragments
// software-pipelined one step ahead so loads hide under the softmax VALU chain.
__global__ void __launch_bounds__(256)
attn_kernel(const unsigned short* __restrict__ qg,
            const unsigned short* __restrict__ kg,
            const unsigned short* __restrict__ vg,
            unsigned short* __restrict__ ob) {
  __shared__ __align__(16) unsigned short plds[8][16 * 32];
  int lane = threadIdx.x & 31;
  int wave = threadIdx.x >> 5;
  int tile = blockIdx.x * 8 + wave;        // 0..4095
  int qt = tile & (SEQ / 16 - 1);          // 0..127
  int bh = tile >> 7;                      // 0..31  (b*NHEAD + h)
  const unsigned short* qp = qg + (size_t)bh * SEQ * HDIM;   // [T, Hd]
  const unsigned short* kp = kg + (size_t)bh * SEQ * HDIM;   // [T, Hd]
  const unsigned short* vp = vg + (size_t)bh * HDIM * SEQ;   // [Hd, T]
  int q0 = qt * 16;
  v16bf aq0 = load_a(qp, HDIM, q0, 0, lane);
  v16bf aq1 = load_a(qp, HDIM, q0, 32, lane);
  v8f o0 = {}, o1 = {}, o2 = {}, o3 = {};
  float mst[8], lst[8];
#pragma unroll
  for (int j = 0; j < 8; ++j) { mst[j] = -1e30f; lst[j] = 0.f; }
  unsigned short* pl = &plds[wave][0];
  int rowadd = (lane >> 4) << 3;
  int col = lane & 15;

  // prime K-fragment pipeline for j0 = 0
  v16bf bk0 = load_b(kp, HDIM, 0, 0, lane);
  v16bf bk1 = load_b(kp, HDIM, 0, 32, lane);
  v16bf bk2 = load_b(kp, HDIM, 16, 0, lane);
  v16bf bk3 = load_b(kp, HDIM, 16, 32, lane);

#pragma unroll 1
  for (int j0 = 0; j0 < SEQ; j0 += 32) {
    v8f s0 = {}, s1 = {};
    s0 = wmma_bf16(aq0, bk0, s0);
    s1 = wmma_bf16(aq0, bk2, s1);
    s0 = wmma_bf16(aq1, bk1, s0);
    s1 = wmma_bf16(aq1, bk3, s1);
    // prefetch next step's K fragments and this step's V fragments now;
    // they are consumed only after the softmax VALU chain below.
    int jn = (j0 + 32) & (SEQ - 1);
    v16bf nk0 = load_b(kp, HDIM, jn, 0, lane);
    v16bf nk1 = load_b(kp, HDIM, jn, 32, lane);
    v16bf nk2 = load_b(kp, HDIM, jn + 16, 0, lane);
    v16bf nk3 = load_b(kp, HDIM, jn + 16, 32, lane);
    v16bf bv0 = load_b(vp, SEQ, 0, j0, lane);
    v16bf bv1 = load_b(vp, SEQ, 16, j0, lane);
    v16bf bv2 = load_b(vp, SEQ, 32, j0, lane);
    v16bf bv3 = load_b(vp, SEQ, 48, j0, lane);
    // online softmax; element j <-> row (rowadd+j), col = key (lane&15).
#pragma unroll
    for (int j = 0; j < 8; ++j) {
      float a0 = s0[j] * 0.125f;
      float a1 = s1[j] * 0.125f;
      float cm = fmaxf(a0, a1);
      cm = fmaxf(cm, __shfl_xor(cm, 1, 32));
      cm = fmaxf(cm, __shfl_xor(cm, 2, 32));
      cm = fmaxf(cm, __shfl_xor(cm, 4, 32));
      cm = fmaxf(cm, __shfl_xor(cm, 8, 32));
      float mn = fmaxf(mst[j], cm);
      float al = __expf(mst[j] - mn);
      float p0 = __expf(a0 - mn);
      float p1 = __expf(a1 - mn);
      float rs = p0 + p1;
      rs += __shfl_xor(rs, 1, 32);
      rs += __shfl_xor(rs, 2, 32);
      rs += __shfl_xor(rs, 4, 32);
      rs += __shfl_xor(rs, 8, 32);
      lst[j] = lst[j] * al + rs;
      mst[j] = mn;
      o0[j] *= al; o1[j] *= al; o2[j] *= al; o3[j] *= al;
      int row = rowadd + j;
      pl[row * 32 + col]      = f2bf(p0);
      pl[row * 32 + 16 + col] = f2bf(p1);
    }
    // re-read P (16x32) as an A fragment from LDS (C-layout -> A-layout).
    Frag pf;
    const unsigned short* pq = pl + (lane & 15) * 32 + ((lane >> 4) << 3);
    pf.u[0] = *(const uint4*)(pq);
    pf.u[1] = *(const uint4*)(pq + 16);
    o0 = wmma_bf16(pf.v, bv0, o0);
    o1 = wmma_bf16(pf.v, bv1, o1);
    o2 = wmma_bf16(pf.v, bv2, o2);
    o3 = wmma_bf16(pf.v, bv3, o3);
    bk0 = nk0; bk1 = nk1; bk2 = nk2; bk3 = nk3;
  }
  int b = bh >> 4;
  int h = bh & (NHEAD - 1);
#pragma unroll
  for (int j = 0; j < 8; ++j) {
    float inv = 1.f / lst[j];
    int t = q0 + rowadd + j;
    size_t base = (size_t)(b * SEQ + t) * DMODEL + h * HDIM + col;
    ob[base +  0] = f2bf(o0[j] * inv);
    ob[base + 16] = f2bf(o1[j] * inv);
    ob[base + 32] = f2bf(o2[j] * inv);
    ob[base + 48] = f2bf(o3[j] * inv);
  }
}

// ---------------- layernorm ----------------
__global__ void ln_kernel(const float* __restrict__ y,
                          const float* __restrict__ g,
                          const float* __restrict__ bln,
                          float* __restrict__ out) {
  int row = blockIdx.x;
  const float* yr = y + (size_t)row * DMODEL;
  int lane = threadIdx.x & 31;
  int wave = threadIdx.x >> 5;
  float s = 0.f, s2 = 0.f;
  for (int i = threadIdx.x; i < DMODEL; i += 256) {
    float v = yr[i];
    s += v;
    s2 += v * v;
  }
#pragma unroll
  for (int m = 16; m >= 1; m >>= 1) {
    s += __shfl_xor(s, m, 32);
    s2 += __shfl_xor(s2, m, 32);
  }
  __shared__ float sred[16];
  if (lane == 0) { sred[wave] = s; sred[8 + wave] = s2; }
  __syncthreads();
  if (threadIdx.x == 0) {
    float ts = 0.f, ts2 = 0.f;
    for (int i = 0; i < 8; ++i) { ts += sred[i]; ts2 += sred[8 + i]; }
    float mu = ts * (1.f / DMODEL);
    float var = ts2 * (1.f / DMODEL) - mu * mu;
    sred[0] = mu;
    sred[1] = rsqrtf(var + 1e-5f);
  }
  __syncthreads();
  float mu = sred[0], rstd = sred[1];
  for (int i = threadIdx.x; i < DMODEL; i += 256)
    out[(size_t)row * DMODEL + i] = (yr[i] - mu) * rstd * g[i] + bln[i];
}

// ---------------- launcher ----------------
extern "C" void kernel_launch(void* const* d_in, const int* in_sizes, int n_in,
                              void* d_out, int out_size, void* d_ws,
                              size_t ws_size, hipStream_t stream) {
  const float* x   = (const float*)d_in[0];
  const float* Wq  = (const float*)d_in[1];
  const float* bq  = (const float*)d_in[2];
  const float* Wk  = (const float*)d_in[3];
  const float* bk  = (const float*)d_in[4];
  const float* Wv  = (const float*)d_in[5];
  const float* bv  = (const float*)d_in[6];
  const float* Wo  = (const float*)d_in[7];
  const float* bo  = (const float*)d_in[8];
  const float* lng = (const float*)d_in[9];
  const float* lnb = (const float*)d_in[10];
  float* out = (float*)d_out;

  char* ws = (char*)d_ws;
  const size_t MB = 1024ull * 1024ull;
  unsigned short* xb  = (unsigned short*)(ws + 0);        //  8 MB  x bf16
  unsigned short* wqT = (unsigned short*)(ws + 8 * MB);   //  2 MB
  unsigned short* wkT = (unsigned short*)(ws + 10 * MB);  //  2 MB
  unsigned short* wvT = (unsigned short*)(ws + 12 * MB);  //  2 MB
  unsigned short* woT = (unsigned short*)(ws + 14 * MB);  //  2 MB
  unsigned short* qh  = (unsigned short*)(ws + 16 * MB);  //  8 MB  [B,H,T,Hd]
  unsigned short* kh  = (unsigned short*)(ws + 24 * MB);  //  8 MB  [B,H,T,Hd]
  unsigned short* vt  = (unsigned short*)(ws + 32 * MB);  //  8 MB  [B,H,Hd,T]
  unsigned short* ob  = (unsigned short*)(ws + 40 * MB);  //  8 MB  [M,D]
  float* ybuf         = (float*)(ws + 48 * MB);           // 16 MB  [M,D]

  dim3 tb(32, 8), tg(DMODEL / 32, DMODEL / 32);
  wtrans_kernel<<<tg, tb, 0, stream>>>(Wq, wqT);
  wtrans_kernel<<<tg, tb, 0, stream>>>(Wk, wkT);
  wtrans_kernel<<<tg, tb, 0, stream>>>(Wv, wvT);
  wtrans_kernel<<<tg, tb, 0, stream>>>(Wo, woT);
  cvt_bf16_kernel<<<4096, 256, 0, stream>>>(x, xb, MROWS * DMODEL);

  dim3 gg(MROWS / 64, DMODEL / 128);
  gemm_tdm_kernel<<<gg, 256, 0, stream>>>(xb, wqT, bq, nullptr, qh, nullptr, 0);
  gemm_tdm_kernel<<<gg, 256, 0, stream>>>(xb, wkT, bk, nullptr, kh, nullptr, 0);
  gemm_tdm_kernel<<<gg, 256, 0, stream>>>(xb, wvT, bv, nullptr, vt, nullptr, 1);

  attn_kernel<<<(NBATCH * NHEAD * (SEQ / 16)) / 8, 256, 0, stream>>>(qh, kh,
                                                                     vt, ob);
  gemm_tdm_kernel<<<gg, 256, 0, stream>>>(ob, woT, bo, x, nullptr, ybuf, 2);
  ln_kernel<<<MROWS, 256, 0, stream>>>(ybuf, lng, lnb, out);
}
